// DGCNN_Local_87256555585566
// MI455X (gfx1250) — matrix operand order repaired and checked by reading, not compiled, not run on hardware
//
#include <hip/hip_runtime.h>

// DGCNN forward for MI455X (gfx1250, wave32).
// All heavy GEMMs (edge-conv u/v, kNN X^T X, conv5) run on
// v_wmma_f32_16x16x32_bf16 (f32 accumulate). Layout per CDNA5 ISA 7.12.2:
//   A (16x32 bf16): lane l (row = l&15): elems0-7 = K kb+8*(l>>4).., elems8-15 = +16
//   B (32x16 bf16): lane l (col = l&15): elems0-7 = K kb+8*(l>>4).., elems8-15 = +16
//   D (16x16 f32) : lane l: N = l&15, VGPR r -> M = r + 8*(l>>4)
// GEMM waves compute 16x64 (4 col tiles) per A fragment to amortize A loads.

#define NB   16
#define NP   1024
#define KNN  20
#define EMBC 1024
#define NOUT 40
#define LDBX 512               // bf16 activation row length (concat channels)
#define BN_INV 0.9999950000374997f   // 1/sqrt(1+1e-5)
#define NEG_INF (-3.402823466e38f)

typedef __attribute__((ext_vector_type(16))) __bf16 v16bf;
typedef __attribute__((ext_vector_type(8)))  float  v8f;

union ABf { uint4 q[2]; v16bf v; };

__device__ __forceinline__ unsigned short f2bf(float f) {
  unsigned u = __float_as_uint(f);
  u += 0x7FFFu + ((u >> 16) & 1u);          // round-to-nearest-even
  return (unsigned short)(u >> 16);
}
__device__ __forceinline__ float lrelu(float v) { return v >= 0.f ? v : 0.2f * v; }

__device__ __forceinline__ v8f wmma_bf16(v16bf a, v16bf b, v8f c) {
  return __builtin_amdgcn_wmma_f32_16x16x32_bf16(false, a, false, b, (short)0, c,
                                                 false, false);
}

// ---------------------------------------------------------------------------
// Kernel 1: local frames. Per point: full 3x3 covariance over all N diffs,
// closed-form symmetric eigendecomposition (ascending), top-20 LARGEST
// squared distances (y_sub==y since diag(y)=0), emit x0[b][6][n][k]
// (channels 3..5 duplicate 0..2).
// ---------------------------------------------------------------------------
__global__ __launch_bounds__(256) void k_local_frames(const float* __restrict__ x,
                                                      float* __restrict__ x0) {
  __shared__ float xs[3 * NP];
  const int b = blockIdx.y;
  for (int i = threadIdx.x; i < 3 * NP; i += 256) xs[i] = x[(size_t)b * 3 * NP + i];
  __syncthreads();
  const int w = threadIdx.x >> 5, l = threadIdx.x & 31;
  const int n = blockIdx.x * 8 + w;
  const float px = xs[n], py = xs[NP + n], pz = xs[2 * NP + n];

  float red[32];
  float cxx = 0, cxy = 0, cxz = 0, cyy = 0, cyz = 0, czz = 0;
  for (int i = 0; i < 32; ++i) {
    int m = l * 32 + i;
    float dx = xs[m] - px, dy = xs[NP + m] - py, dz = xs[2 * NP + m] - pz;
    cxx += dx * dx; cxy += dx * dy; cxz += dx * dz;
    cyy += dy * dy; cyz += dy * dz; czz += dz * dz;
    red[i] = dx * dx + dy * dy + dz * dz;
  }
  for (int off = 16; off; off >>= 1) {
    cxx += __shfl_xor(cxx, off); cxy += __shfl_xor(cxy, off);
    cxz += __shfl_xor(cxz, off); cyy += __shfl_xor(cyy, off);
    cyz += __shfl_xor(cyz, off); czz += __shfl_xor(czz, off);
  }

  // --- analytic 3x3 symmetric eigendecomposition, ascending eigenvalues ---
  float V[3][3];
  {
    const float a00 = cxx, a01 = cxy, a02 = cxz, a11 = cyy, a12 = cyz, a22 = czz;
    const float q  = (a00 + a11 + a22) * (1.0f / 3.0f);
    const float d0 = a00 - q, d1 = a11 - q, d2 = a22 - q;
    const float p1 = a01 * a01 + a02 * a02 + a12 * a12;
    const float p2 = d0 * d0 + d1 * d1 + d2 * d2 + 2.0f * p1;
    const float p  = sqrtf(p2 * (1.0f / 6.0f)) + 1e-20f;
    const float ip = 1.0f / p;
    const float b00 = d0 * ip, b01 = a01 * ip, b02 = a02 * ip;
    const float b11 = d1 * ip, b12 = a12 * ip, b22 = d2 * ip;
    float detB = b00 * (b11 * b22 - b12 * b12) - b01 * (b01 * b22 - b12 * b02)
               + b02 * (b01 * b12 - b11 * b02);
    float r = fminf(1.0f, fmaxf(-1.0f, 0.5f * detB));
    float phi = acosf(r) * (1.0f / 3.0f);
    float e2 = q + 2.0f * p * cosf(phi);
    float e0 = q + 2.0f * p * cosf(phi + 2.0943951023931953f);

    auto evec = [&](float lam, float out[3]) {
      float r0x = a00 - lam, r0y = a01,       r0z = a02;
      float r1x = a01,       r1y = a11 - lam, r1z = a12;
      float r2x = a02,       r2y = a12,       r2z = a22 - lam;
      float c0x = r0y * r1z - r0z * r1y, c0y = r0z * r1x - r0x * r1z, c0z = r0x * r1y - r0y * r1x;
      float c1x = r0y * r2z - r0z * r2y, c1y = r0z * r2x - r0x * r2z, c1z = r0x * r2y - r0y * r2x;
      float c2x = r1y * r2z - r1z * r2y, c2y = r1z * r2x - r1x * r2z, c2z = r1x * r2y - r1y * r2x;
      float n0 = c0x * c0x + c0y * c0y + c0z * c0z;
      float n1 = c1x * c1x + c1y * c1y + c1z * c1z;
      float n2 = c2x * c2x + c2y * c2y + c2z * c2z;
      float vx = c0x, vy = c0y, vz = c0z, nn = n0;
      if (n1 > nn) { vx = c1x; vy = c1y; vz = c1z; nn = n1; }
      if (n2 > nn) { vx = c2x; vy = c2y; vz = c2z; nn = n2; }
      if (nn < 1e-30f) { vx = 1.f; vy = 0.f; vz = 0.f; nn = 1.f; }
      float s = rsqrtf(nn);
      out[0] = vx * s; out[1] = vy * s; out[2] = vz * s;
    };
    float v0[3], v2[3];
    evec(e0, v0);
    evec(e2, v2);
    float v1[3] = { v2[1] * v0[2] - v2[2] * v0[1],
                    v2[2] * v0[0] - v2[0] * v0[2],
                    v2[0] * v0[1] - v2[1] * v0[0] };
    float nn1 = v1[0] * v1[0] + v1[1] * v1[1] + v1[2] * v1[2];
    float s1 = (nn1 > 1e-30f) ? rsqrtf(nn1) : 0.f;
    for (int d = 0; d < 3; ++d) { V[d][0] = v0[d]; V[d][1] = v1[d] * s1; V[d][2] = v2[d]; }
  }

  // --- top-20 largest red, jax tie-break (lowest index first) ---
  int selm = 0;
  for (int t = 0; t < KNN; ++t) {
    float bv = NEG_INF; int bi = 0;
    for (int i = 0; i < 32; ++i) { float v = red[i]; if (v > bv) { bv = v; bi = l * 32 + i; } }
    for (int off = 16; off; off >>= 1) {
      float ov = __shfl_xor(bv, off); int oi = __shfl_xor(bi, off);
      if (ov > bv || (ov == bv && oi < bi)) { bv = ov; bi = oi; }
    }
    if (l == t) selm = bi;
    if ((bi >> 5) == l) red[bi & 31] = NEG_INF;
  }
  if (l < KNN) {
    int m = selm;
    float dx = xs[m] - px, dy = xs[NP + m] - py, dz = xs[2 * NP + m] - pz;
    for (int c = 0; c < 3; ++c) {
      float y = V[0][c] * dx + V[1][c] * dy + V[2][c] * dz;
      x0[(((size_t)b * 6 + c) * NP + n) * KNN + l]     = y;
      x0[(((size_t)b * 6 + c + 3) * NP + n) * KNN + l] = y;
    }
  }
}

// ---------------------------------------------------------------------------
// conv1 (6->64) + BN + LReLU + max over k  -> x1[b][64][n]
// ---------------------------------------------------------------------------
__global__ __launch_bounds__(256) void k_conv1(const float* __restrict__ x0,
                                               const float* __restrict__ w1,
                                               const float* __restrict__ g,
                                               const float* __restrict__ bb,
                                               float* __restrict__ x1) {
  const int n = blockIdx.x * 256 + threadIdx.x;
  const int o = blockIdx.y, b = blockIdx.z;
  float wv[6];
  for (int c = 0; c < 6; ++c) wv[c] = w1[o * 6 + c];
  const float s = BN_INV * g[o], t = bb[o];
  float m = NEG_INF;
  for (int k = 0; k < KNN; ++k) {
    float h = 0.f;
    for (int c = 0; c < 6; ++c) h += wv[c] * x0[(((size_t)b * 6 + c) * NP + n) * KNN + k];
    m = fmaxf(m, lrelu(h * s + t));
  }
  x1[((size_t)b * 64 + o) * NP + n] = m;
}

// f32 activation [b][C][n] -> bf16 concat buffer xcbf[(b*NP+n)*512 + coff + c]
__global__ __launch_bounds__(256) void k_cvt_act(const float* __restrict__ xf, int C,
                                                 int coff, unsigned short* __restrict__ xcbf) {
  const int n = blockIdx.x * 256 + threadIdx.x;
  const int c = blockIdx.y, b = blockIdx.z;
  xcbf[((size_t)(b * NP + n)) * LDBX + coff + c] = f2bf(xf[((size_t)b * C + c) * NP + n]);
}

// per-point squared norm
__global__ __launch_bounds__(256) void k_xx(const float* __restrict__ xf, int C,
                                            float* __restrict__ xx) {
  const int n = blockIdx.x * 256 + threadIdx.x, b = blockIdx.y;
  float s = 0.f;
  for (int c = 0; c < C; ++c) { float v = xf[((size_t)b * C + c) * NP + n]; s += v * v; }
  xx[b * NP + n] = s;
}

// split W (O x 2C) into bf16 wA = W[:, :C] and wV = W[:, C:] - W[:, :C]
__global__ void k_cvt_wuv(const float* __restrict__ w, int C,
                          unsigned short* __restrict__ wA, unsigned short* __restrict__ wV) {
  const int c = threadIdx.x, o = blockIdx.x;
  float a  = w[(size_t)o * 2 * C + c];
  float b2 = w[(size_t)o * 2 * C + C + c];
  wA[(size_t)o * C + c] = f2bf(a);
  wV[(size_t)o * C + c] = f2bf(b2 - a);
}

// plain f32 -> bf16 weight convert (w5: 1024 x 512)
__global__ void k_cvt_w(const float* __restrict__ w, unsigned short* __restrict__ o5) {
  const int c = threadIdx.x, o = blockIdx.x;
  o5[(size_t)o * 512 + c] = f2bf(w[(size_t)o * 512 + c]);
}

// ---------------------------------------------------------------------------
// kNN: WMMA bf16 X^T X tiles -> pd strip in LDS -> fused top-20 per row.
// grid (NP/16, NB), 8 waves. pd = 2*S - xx[n] - xx[m]; top_k(pd) largest.
// K-step outer loop: one A fragment feeds all 8 m-tiles of the wave.
// ---------------------------------------------------------------------------
__global__ __launch_bounds__(256) void k_knn(const unsigned short* __restrict__ xbf,
                                             int coff, int C,
                                             const float* __restrict__ xx,
                                             int* __restrict__ idxo) {
  __shared__ float pd[16 * NP];          // 64 KB
  const int b = blockIdx.y;
  const int n0 = blockIdx.x * 16;
  const int w = threadIdx.x >> 5, l = threadIdx.x & 31;
  const int ll = l & 15, lh = l >> 4;
  const size_t arow = ((size_t)(b * NP + n0 + ll)) * LDBX + coff;
  float xxn[8];
  for (int r = 0; r < 8; ++r) xxn[r] = xx[b * NP + n0 + 8 * lh + r];

  v8f acc[8] = {};
  for (int kb = 0; kb < C; kb += 32) {
    ABf a;
    const unsigned short* ap = xbf + arow + kb + lh * 8;
    a.q[0] = *(const uint4*)(ap);
    a.q[1] = *(const uint4*)(ap + 16);
#pragma unroll
    for (int jj = 0; jj < 8; ++jj) {
      const int m0 = (w + 8 * jj) * 16;
      ABf bm;
      const unsigned short* bp =
          xbf + ((size_t)(b * NP + m0 + ll)) * LDBX + coff + kb + lh * 8;
      bm.q[0] = *(const uint4*)(bp);
      bm.q[1] = *(const uint4*)(bp + 16);
      acc[jj] = wmma_bf16(a.v, bm.v, acc[jj]);
    }
  }
#pragma unroll
  for (int jj = 0; jj < 8; ++jj) {
    const int m0 = (w + 8 * jj) * 16;
    const float xxm = xx[b * NP + m0 + ll];
    for (int r = 0; r < 8; ++r)
      pd[(8 * lh + r) * NP + m0 + ll] = 2.0f * acc[jj][r] - xxn[r] - xxm;
  }
  __syncthreads();

  for (int rr = 2 * w; rr < 2 * w + 2; ++rr) {
    float vals[32];
    const int base = rr * NP + l * 32;
    for (int i = 0; i < 32; ++i) vals[i] = pd[base + i];
    int* op = idxo + ((size_t)(b * NP + n0 + rr)) * KNN;
    for (int t = 0; t < KNN; ++t) {
      float bv = NEG_INF; int bi = 0;
      for (int i = 0; i < 32; ++i) { if (vals[i] > bv) { bv = vals[i]; bi = l * 32 + i; } }
      for (int off = 16; off; off >>= 1) {
        float ov = __shfl_xor(bv, off); int oi = __shfl_xor(bi, off);
        if (ov > bv || (ov == bv && oi < bi)) { bv = ov; bi = oi; }
      }
      if (l == 0) op[t] = bi;
      if ((bi >> 5) == l) vals[bi & 31] = NEG_INF;
    }
  }
}

// ---------------------------------------------------------------------------
// Generic bf16 WMMA GEMM: out[b][o][n] = sum_c A[o][c] * Xbf[(b*NP+n)*512+coff+c]
// one wave per 16x64 strip (4 col tiles share each A fragment);
// grid (NB*NP/64, O/16), block 32.
// ---------------------------------------------------------------------------
__global__ __launch_bounds__(32) void k_gemm(const unsigned short* __restrict__ A,
                                             const unsigned short* __restrict__ Xbf,
                                             int coff, int C, int O,
                                             float* __restrict__ out) {
  const int l = threadIdx.x, ll = l & 15, lh = l >> 4;
  const int ct0 = blockIdx.x * 4, ot = blockIdx.y;
  const size_t arow = (size_t)(ot * 16 + ll) * C;
  const size_t brow0 = (size_t)(ct0 * 16 + ll) * LDBX + coff;
  v8f acc[4] = {};
  for (int kb = 0; kb < C; kb += 32) {
    ABf a;
    const unsigned short* ap = A + arow + kb + lh * 8;
    a.q[0] = *(const uint4*)(ap);
    a.q[1] = *(const uint4*)(ap + 16);
#pragma unroll
    for (int t = 0; t < 4; ++t) {
      ABf bm;
      const unsigned short* bp = Xbf + brow0 + (size_t)t * 16 * LDBX + kb + lh * 8;
      bm.q[0] = *(const uint4*)(bp);
      bm.q[1] = *(const uint4*)(bp + 16);
      acc[t] = wmma_bf16(a.v, bm.v, acc[t]);
    }
  }
  const int b = ct0 >> 6;
#pragma unroll
  for (int t = 0; t < 4; ++t) {
    const int n = (((ct0 + t) & 63) << 4) + ll;
    for (int r = 0; r < 8; ++r) {
      int o = ot * 16 + 8 * lh + r;
      out[((size_t)b * O + o) * NP + n] = acc[t][r];
    }
  }
}

// ---------------------------------------------------------------------------
// conv5 GEMM (C=512) + BN + LReLU fused with deterministic per-tile max/sum
// partials. One wave per 16x64 strip; grid (NB*NP/64, EMBC/16).
// ---------------------------------------------------------------------------
__global__ __launch_bounds__(32) void k_conv5(const unsigned short* __restrict__ A,
                                              const unsigned short* __restrict__ Xbf,
                                              const float* __restrict__ g,
                                              const float* __restrict__ bb,
                                              float* __restrict__ pmax,
                                              float* __restrict__ psum) {
  const int l = threadIdx.x, ll = l & 15, lh = l >> 4;
  const int ct0 = blockIdx.x * 4, ot = blockIdx.y;
  const size_t arow = (size_t)(ot * 16 + ll) * LDBX;
  const size_t brow0 = (size_t)(ct0 * 16 + ll) * LDBX;
  v8f acc[4] = {};
  for (int kb = 0; kb < LDBX; kb += 32) {
    ABf a;
    const unsigned short* ap = A + arow + kb + lh * 8;
    a.q[0] = *(const uint4*)(ap);
    a.q[1] = *(const uint4*)(ap + 16);
#pragma unroll
    for (int t = 0; t < 4; ++t) {
      ABf bm;
      const unsigned short* bp = Xbf + brow0 + (size_t)t * 16 * LDBX + kb + lh * 8;
      bm.q[0] = *(const uint4*)(bp);
      bm.q[1] = *(const uint4*)(bp + 16);
      acc[t] = wmma_bf16(a.v, bm.v, acc[t]);
    }
  }
  const int b = ct0 >> 6;
#pragma unroll
  for (int t = 0; t < 4; ++t) {
    const int tcol = (ct0 + t) & 63;
    for (int r = 0; r < 8; ++r) {
      int o = ot * 16 + 8 * lh + r;
      float val = lrelu(acc[t][r] * (BN_INV * g[o]) + bb[o]);
      float vmax = val, vsum = val;
      for (int off = 8; off; off >>= 1) {     // reduce 16 lanes sharing o
        vmax = fmaxf(vmax, __shfl_xor(vmax, off));
        vsum += __shfl_xor(vsum, off);
      }
      if (ll == 0) {
        pmax[((size_t)(b * EMBC + o)) * 64 + tcol] = vmax;
        psum[((size_t)(b * EMBC + o)) * 64 + tcol] = vsum;
      }
    }
  }
}

__global__ __launch_bounds__(256) void k_pool(const float* __restrict__ pmax,
                                              const float* __restrict__ psum,
                                              float* __restrict__ fmax,
                                              float* __restrict__ fsum) {
  const int i = blockIdx.x * 256 + threadIdx.x;   // i = b*EMBC + o
  float m = NEG_INF, s = 0.f;
  for (int t = 0; t < 64; ++t) {
    m = fmaxf(m, pmax[(size_t)i * 64 + t]);
    s += psum[(size_t)i * 64 + t];
  }
  fmax[i] = m; fsum[i] = s;
}

// ---------------------------------------------------------------------------
// edge gather + BN + LReLU + max over k: x[b][o][n] = max_k act(u[o,idx]+v[o,n])
// ---------------------------------------------------------------------------
__global__ __launch_bounds__(256) void k_edge_max(const float* __restrict__ u,
                                                  const float* __restrict__ v,
                                                  const int* __restrict__ idx, int O,
                                                  const float* __restrict__ g,
                                                  const float* __restrict__ bb,
                                                  float* __restrict__ xo) {
  const int n = blockIdx.x * 256 + threadIdx.x;
  const int o = blockIdx.y, b = blockIdx.z;
  const float vn = v[((size_t)b * O + o) * NP + n];
  const float s = BN_INV * g[o], t = bb[o];
  const int* ip = idx + ((size_t)(b * NP + n)) * KNN;
  const float* ub = u + ((size_t)b * O + o) * NP;
  float m = NEG_INF;
  for (int k = 0; k < KNN; ++k) m = fmaxf(m, lrelu((ub[ip[k]] + vn) * s + t));
  xo[((size_t)b * O + o) * NP + n] = m;
}

// ---------------------------------------------------------------------------
// tiny FC head (B=16 rows -> plain VALU)
// ---------------------------------------------------------------------------
__global__ __launch_bounds__(512) void k_fc1(const float* __restrict__ fmax,
                                             const float* __restrict__ fsum,
                                             const float* __restrict__ lw1,
                                             const float* __restrict__ g,
                                             const float* __restrict__ bb,
                                             float* __restrict__ f1) {
  const int o = threadIdx.x, b = blockIdx.x;
  const float* wr = lw1 + (size_t)o * (2 * EMBC);
  float s = 0.f;
  for (int j = 0; j < EMBC; ++j) s += fmax[b * EMBC + j] * wr[j];
  const float invN = 1.0f / NP;
  for (int j = 0; j < EMBC; ++j) s += (fsum[b * EMBC + j] * invN) * wr[EMBC + j];
  f1[b * 512 + o] = lrelu(s * (BN_INV * g[o]) + bb[o]);
}

__global__ __launch_bounds__(256) void k_fc2(const float* __restrict__ f1,
                                             const float* __restrict__ lw2,
                                             const float* __restrict__ lb2,
                                             const float* __restrict__ g,
                                             const float* __restrict__ bb,
                                             float* __restrict__ f2) {
  const int o = threadIdx.x, b = blockIdx.x;
  const float* wr = lw2 + (size_t)o * 512;
  float s = lb2[o];
  for (int j = 0; j < 512; ++j) s += f1[b * 512 + j] * wr[j];
  f2[b * 256 + o] = lrelu(s * (BN_INV * g[o]) + bb[o]);
}

__global__ __launch_bounds__(64) void k_fc3(const float* __restrict__ f2,
                                            const float* __restrict__ lw3,
                                            const float* __restrict__ lb3,
                                            float* __restrict__ out) {
  const int o = threadIdx.x, b = blockIdx.x;
  if (o < NOUT) {
    const float* wr = lw3 + (size_t)o * 256;
    float s = lb3[o];
    for (int j = 0; j < 256; ++j) s += f2[b * 256 + j] * wr[j];
    out[b * NOUT + o] = s;
  }
}

// ---------------------------------------------------------------------------
extern "C" void kernel_launch(void* const* d_in, const int* in_sizes, int n_in,
                              void* d_out, int out_size, void* d_ws, size_t ws_size,
                              hipStream_t stream) {
  (void)in_sizes; (void)n_in; (void)out_size; (void)ws_size;
  const float* x   = (const float*)d_in[0];
  const float* w1  = (const float*)d_in[1];
  const float* g1  = (const float*)d_in[2];
  const float* b1  = (const float*)d_in[3];
  const float* w2  = (const float*)d_in[4];
  const float* g2  = (const float*)d_in[5];
  const float* b2  = (const float*)d_in[6];
  const float* w3  = (const float*)d_in[7];
  const float* g3  = (const float*)d_in[8];
  const float* b3  = (const float*)d_in[9];
  const float* w4  = (const float*)d_in[10];
  const float* g4  = (const float*)d_in[11];
  const float* b4  = (const float*)d_in[12];
  const float* w5  = (const float*)d_in[13];
  const float* g5  = (const float*)d_in[14];
  const float* b5  = (const float*)d_in[15];
  const float* lw1 = (const float*)d_in[16];
  const float* g6  = (const float*)d_in[17];
  const float* b6  = (const float*)d_in[18];
  const float* lw2 = (const float*)d_in[19];
  const float* lb2 = (const float*)d_in[20];
  const float* g7  = (const float*)d_in[21];
  const float* b7  = (const float*)d_in[22];
  const float* lw3 = (const float*)d_in[23];
  const float* lb3 = (const float*)d_in[24];
  float* out = (float*)d_out;

  char* base = (char*)d_ws;
  size_t off = 0;
  auto alloc = [&](size_t bytes) -> void* {
    void* p = base + off;
    off = (off + bytes + 255) & ~(size_t)255;
    return p;
  };
  float* x0f = (float*)alloc((size_t)NB * 6 * NP * KNN * 4);      // 7.9 MB
  float* x1f = (float*)alloc((size_t)NB * 64 * NP * 4);           // 4 MB
  float* x2f = (float*)alloc((size_t)NB * 64 * NP * 4);           // 4 MB
  float* x3f = (float*)alloc((size_t)NB * 128 * NP * 4);          // 8 MB
  float* x4f = (float*)alloc((size_t)NB * 256 * NP * 4);          // 16 MB
  unsigned short* xcbf = (unsigned short*)alloc((size_t)NB * NP * LDBX * 2); // 16 MB
  float* xxb = (float*)alloc((size_t)NB * NP * 4);
  int*   idxb = (int*)alloc((size_t)NB * NP * KNN * 4);
  float* uf = (float*)alloc((size_t)NB * 256 * NP * 4);           // 16 MB
  float* vf = (float*)alloc((size_t)NB * 256 * NP * 4);           // 16 MB
  unsigned short* wAb = (unsigned short*)alloc((size_t)256 * 128 * 2);
  unsigned short* wVb = (unsigned short*)alloc((size_t)256 * 128 * 2);
  unsigned short* w5b = (unsigned short*)alloc((size_t)EMBC * 512 * 2);
  float* pmaxb = (float*)alloc((size_t)NB * EMBC * 64 * 4);       // 4 MB
  float* psumb = (float*)alloc((size_t)NB * EMBC * 64 * 4);       // 4 MB
  float* fmaxb = (float*)alloc((size_t)NB * EMBC * 4);
  float* fsumb = (float*)alloc((size_t)NB * EMBC * 4);
  float* f1b = (float*)alloc((size_t)NB * 512 * 4);
  float* f2b = (float*)alloc((size_t)NB * 256 * 4);

  const int CT4 = NB * NP / 64;  // 256 strips of 4 column tiles

  // stage 0: local frames + conv1
  k_local_frames<<<dim3(NP / 8, NB), 256, 0, stream>>>(x, x0f);
  k_conv1<<<dim3(NP / 256, 64, NB), 256, 0, stream>>>(x0f, w1, g1, b1, x1f);
  k_cvt_act<<<dim3(NP / 256, 64, NB), 256, 0, stream>>>(x1f, 64, 0, xcbf);
  k_xx<<<dim3(NP / 256, NB), 256, 0, stream>>>(x1f, 64, xxb);

  // layer 2: knn(x1) -> conv(64 out, 64 in)
  k_knn<<<dim3(NP / 16, NB), 256, 0, stream>>>(xcbf, 0, 64, xxb, idxb);
  k_cvt_wuv<<<64, 64, 0, stream>>>(w2, 64, wAb, wVb);
  k_gemm<<<dim3(CT4, 4), 32, 0, stream>>>(wAb, xcbf, 0, 64, 64, uf);
  k_gemm<<<dim3(CT4, 4), 32, 0, stream>>>(wVb, xcbf, 0, 64, 64, vf);
  k_edge_max<<<dim3(NP / 256, 64, NB), 256, 0, stream>>>(uf, vf, idxb, 64, g2, b2, x2f);
  k_cvt_act<<<dim3(NP / 256, 64, NB), 256, 0, stream>>>(x2f, 64, 64, xcbf);
  k_xx<<<dim3(NP / 256, NB), 256, 0, stream>>>(x2f, 64, xxb);

  // layer 3: knn(x2) -> conv(128 out, 64 in)
  k_knn<<<dim3(NP / 16, NB), 256, 0, stream>>>(xcbf, 64, 64, xxb, idxb);
  k_cvt_wuv<<<128, 64, 0, stream>>>(w3, 64, wAb, wVb);
  k_gemm<<<dim3(CT4, 8), 32, 0, stream>>>(wAb, xcbf, 64, 64, 128, uf);
  k_gemm<<<dim3(CT4, 8), 32, 0, stream>>>(wVb, xcbf, 64, 64, 128, vf);
  k_edge_max<<<dim3(NP / 256, 128, NB), 256, 0, stream>>>(uf, vf, idxb, 128, g3, b3, x3f);
  k_cvt_act<<<dim3(NP / 256, 128, NB), 256, 0, stream>>>(x3f, 128, 128, xcbf);
  k_xx<<<dim3(NP / 256, NB), 256, 0, stream>>>(x3f, 128, xxb);

  // layer 4: knn(x3) -> conv(256 out, 128 in)
  k_knn<<<dim3(NP / 16, NB), 256, 0, stream>>>(xcbf, 128, 128, xxb, idxb);
  k_cvt_wuv<<<256, 128, 0, stream>>>(w4, 128, wAb, wVb);
  k_gemm<<<dim3(CT4, 16), 32, 0, stream>>>(wAb, xcbf, 128, 128, 256, uf);
  k_gemm<<<dim3(CT4, 16), 32, 0, stream>>>(wVb, xcbf, 128, 128, 256, vf);
  k_edge_max<<<dim3(NP / 256, 256, NB), 256, 0, stream>>>(uf, vf, idxb, 256, g4, b4, x4f);
  k_cvt_act<<<dim3(NP / 256, 256, NB), 256, 0, stream>>>(x4f, 256, 256, xcbf);

  // conv5 (1024 x 512) fused BN+LReLU+pool partials, then deterministic pool
  k_cvt_w<<<EMBC, 512, 0, stream>>>(w5, w5b);
  k_conv5<<<dim3(CT4, EMBC / 16), 32, 0, stream>>>(w5b, xcbf, g5, b5, pmaxb, psumb);
  k_pool<<<NB * EMBC / 256, 256, 0, stream>>>(pmaxb, psumb, fmaxb, fsumb);

  // FC head
  k_fc1<<<NB, 512, 0, stream>>>(fmaxb, fsumb, lw1, g6, b6, f1b);
  k_fc2<<<NB, 256, 0, stream>>>(f1b, lw2, lb2, g7, b7, f2b);
  k_fc3<<<NB, 64, 0, stream>>>(f2b, lw3, lb3, out);
}
